// Linear4bit_86440511799690
// MI455X (gfx1250) — matrix operand order repaired and checked by reading, not compiled, not run on hardware
//
#include <hip/hip_runtime.h>

typedef _Float16 v16h __attribute__((ext_vector_type(16)));
typedef _Float16 v8h  __attribute__((ext_vector_type(8)));
typedef float    v8f  __attribute__((ext_vector_type(8)));
typedef int      v4i_vs __attribute__((vector_size(16)));   // matches async builtin param type

#define TOKENS 8192
#define IN_F   4096
#define OUT_F  11008

#define BM 128
#define BN 128

#if defined(__has_builtin)
#  if __has_builtin(__builtin_amdgcn_global_load_async_to_lds_b128)
#    define HAVE_ASYNC_LDS 1
#  endif
#  if __has_builtin(__builtin_amdgcn_s_wait_asynccnt)
#    define HAVE_WAIT_ASYNC 1
#  endif
#endif
#ifndef HAVE_ASYNC_LDS
#  define HAVE_ASYNC_LDS 0
#endif
#ifndef HAVE_WAIT_ASYNC
#  define HAVE_WAIT_ASYNC 0
#endif

#if HAVE_ASYNC_LDS
#define AS1P(p) ((__attribute__((address_space(1))) v4i_vs*)(p))
#define AS3P(p) ((__attribute__((address_space(3))) v4i_vs*)(p))
#endif

__device__ __forceinline__ void async_wait_le8() {
#if HAVE_ASYNC_LDS
#  if HAVE_WAIT_ASYNC
    __builtin_amdgcn_s_wait_asynccnt(8);
#  else
    asm volatile("s_wait_asynccnt 0x8" ::: "memory");
#  endif
#endif
}
__device__ __forceinline__ void async_wait_0() {
#if HAVE_ASYNC_LDS
#  if HAVE_WAIT_ASYNC
    __builtin_amdgcn_s_wait_asynccnt(0);
#  else
    asm volatile("s_wait_asynccnt 0x0" ::: "memory");
#  endif
#endif
}

// ============================================================================
// Pre-pass 1: x fp32 -> f16 (bandwidth bound)
// ============================================================================
__global__ __launch_bounds__(256) void cvt_x_f16(
    const float* __restrict__ x, _Float16* __restrict__ xh)
{
    size_t base = ((size_t)blockIdx.x * 256 + threadIdx.x) * 8;
    float4 a = *reinterpret_cast<const float4*>(x + base);
    float4 b = *reinterpret_cast<const float4*>(x + base + 4);
    _Float16 h[8];
    h[0] = (_Float16)a.x; h[1] = (_Float16)a.y; h[2] = (_Float16)a.z; h[3] = (_Float16)a.w;
    h[4] = (_Float16)b.x; h[5] = (_Float16)b.y; h[6] = (_Float16)b.z; h[7] = (_Float16)b.w;
    *reinterpret_cast<int4*>(xh + base) = *reinterpret_cast<int4*>(h);
}

// ============================================================================
// Pre-pass 2: dequant qweight once: wh = f16((q-8)*scale_row)  (reference math)
// One block per output row; 256 threads x 16 elements = 4096.
// ============================================================================
__global__ __launch_bounds__(256) void dequant_w_f16(
    const int* __restrict__ qweight, const float* __restrict__ scale,
    _Float16* __restrict__ wh)
{
    const int o = blockIdx.x;
    const float s = scale[o];
    const int*  qrow = &qweight[(size_t)o * IN_F];
    _Float16*   wrow = &wh[(size_t)o * IN_F];
    const int k = threadIdx.x * 16;

    int qa[16];
    *reinterpret_cast<int4*>(qa + 0)  = *reinterpret_cast<const int4*>(qrow + k + 0);
    *reinterpret_cast<int4*>(qa + 4)  = *reinterpret_cast<const int4*>(qrow + k + 4);
    *reinterpret_cast<int4*>(qa + 8)  = *reinterpret_cast<const int4*>(qrow + k + 8);
    *reinterpret_cast<int4*>(qa + 12) = *reinterpret_cast<const int4*>(qrow + k + 12);
    _Float16 h[16];
#pragma unroll
    for (int e = 0; e < 16; ++e) h[e] = (_Float16)((float)(qa[e] - 8) * s);
    *reinterpret_cast<int4*>(wrow + k + 0) = *reinterpret_cast<int4*>(h + 0);
    *reinterpret_cast<int4*>(wrow + k + 8) = *reinterpret_cast<int4*>(h + 8);
}

// ============================================================================
// Main GEMM: pure f16 tiles, async->LDS double buffered, zero inner-loop cvt.
// Staged K-tile = 64 (two 16x16x32 sub-steps -> 16 WMMAs/wave per barrier pair)
// ============================================================================
#define KT  64    // staged K per tile
#define LDH 72    // padded LDS row stride in halves (144B = 36 banks, conflict-free)

__global__ __launch_bounds__(256) void gemm_f16_wmma(
    const _Float16* __restrict__ xh,   // [TOKENS][IN_F]
    const _Float16* __restrict__ wh,   // [OUT_F][IN_F]
    const float* __restrict__ bias,
    float* __restrict__ out)
{
    __shared__ __align__(16) _Float16 Ash[2][BM * LDH];
    __shared__ __align__(16) _Float16 Bsh[2][BN * LDH];

    const int tid  = threadIdx.x;
    const int lane = tid & 31;
    const int wave = tid >> 5;

    const int n0 = blockIdx.x * BN;
    const int m0 = blockIdx.y * BM;

    const int wave_m  = wave >> 2;   // 0..1 : 64-row slab of M
    const int wave_n  = wave & 3;    // 0..3 : 32-col slab of N
    const int halfsel = lane >> 4;
    const int l16     = lane & 15;

    v8f acc[4][2] = {};

    // one wave issues 8 async b128 copies per staged tile (4 A + 4 B)
    auto stage = [&](int kt, int buf) {
#pragma unroll
        for (int i = 0; i < 4; ++i) {
            int idx = tid + i * 256;        // 1024 16B chunks per 16KB tile
            int row = idx >> 3;
            int c8  = (idx & 7) * 8;        // halves
            const _Float16* gA = &xh[(size_t)(m0 + row) * IN_F + kt + c8];
            _Float16*       lA = &Ash[buf][row * LDH + c8];
            const _Float16* gB = &wh[(size_t)(n0 + row) * IN_F + kt + c8];
            _Float16*       lB = &Bsh[buf][row * LDH + c8];
#if HAVE_ASYNC_LDS
            __builtin_amdgcn_global_load_async_to_lds_b128(AS1P(gA), AS3P(lA), 0, 0);
            __builtin_amdgcn_global_load_async_to_lds_b128(AS1P(gB), AS3P(lB), 0, 0);
#else
            *reinterpret_cast<int4*>(lA) = *reinterpret_cast<const int4*>(gA);
            *reinterpret_cast<int4*>(lB) = *reinterpret_cast<const int4*>(gB);
#endif
        }
    };

    const int NK = IN_F / KT;   // 64 staged tiles

    stage(0, 0);

    for (int t = 0; t < NK; ++t) {
        const int cur = t & 1;

        __syncthreads();                    // buf[1-cur] readers (iter t-1) done

        if (t + 1 < NK) {
            stage((t + 1) * KT, 1 - cur);   // 8 new async ops in flight
            async_wait_le8();               // tile t's 8 older ops complete
        } else {
            async_wait_0();
        }

        // L2 prefetch two tiles ahead (global_prefetch_b8)
        if (t + 2 < NK) {
            int row = tid >> 3, c8 = (tid & 7) * 8;
            __builtin_prefetch(&xh[(size_t)(m0 + row) * IN_F + (t + 2) * KT + c8], 0, 2);
            __builtin_prefetch(&wh[(size_t)(n0 + row) * IN_F + (t + 2) * KT + c8], 0, 2);
        }

        __syncthreads();                    // tile t visible from every wave

#pragma unroll
        for (int ks = 0; ks < KT; ks += 32) {
            // A fragments (16x32 f16, ISA 7.12.2): lanes 0-15 K{0-7,16-23}, 16-31 K{8-15,24-31}
            v16h afrag[4];
#pragma unroll
            for (int mi = 0; mi < 4; ++mi) {
                const _Float16* base =
                    &Ash[cur][(wave_m * 64 + mi * 16 + l16) * LDH + ks + halfsel * 8];
                const v8h lo = *reinterpret_cast<const v8h*>(base);
                const v8h hi = *reinterpret_cast<const v8h*>(base + 16);
                afrag[mi] = __builtin_shufflevector(lo, hi,
                    0,1,2,3,4,5,6,7,8,9,10,11,12,13,14,15);
            }
            // B fragments (32x16 f16): lanes 0-15 K{0-7,16-23}, 16-31 K{8-15,24-31}
            v16h bfrag[2];
#pragma unroll
            for (int ni = 0; ni < 2; ++ni) {
                const _Float16* base =
                    &Bsh[cur][(wave_n * 32 + ni * 16 + l16) * LDH + ks + halfsel * 8];
                const v8h lo = *reinterpret_cast<const v8h*>(base);
                const v8h hi = *reinterpret_cast<const v8h*>(base + 16);
                bfrag[ni] = __builtin_shufflevector(lo, hi,
                    0,1,2,3,4,5,6,7,8,9,10,11,12,13,14,15);
            }
#pragma unroll
            for (int mi = 0; mi < 4; ++mi)
#pragma unroll
                for (int ni = 0; ni < 2; ++ni)
                    acc[mi][ni] = __builtin_amdgcn_wmma_f32_16x16x32_f16(
                        false, afrag[mi], false, bfrag[ni],
                        (short)0, acc[mi][ni], false, false);
        }
    }

    // epilogue: bias add + store (C layout: VGPR r -> rows r / r+8)
#pragma unroll
    for (int ni = 0; ni < 2; ++ni) {
        int col = n0 + wave_n * 32 + ni * 16 + l16;
        float bv = bias[col];
#pragma unroll
        for (int mi = 0; mi < 4; ++mi) {
            int rbase = m0 + wave_m * 64 + mi * 16 + halfsel * 8;
#pragma unroll
            for (int r = 0; r < 8; ++r) {
                out[(size_t)(rbase + r) * OUT_F + col] = acc[mi][ni][r] + bv;
            }
        }
    }
}

// ============================================================================
// Fallback: fused single-kernel variant (used only if d_ws is too small).
// ============================================================================
#define BK  32
#define LDR 36

__global__ __launch_bounds__(256) void linear4bit_wmma_fused(
    const float* __restrict__ x,
    const int*   __restrict__ qweight,
    const float* __restrict__ scale,
    const float* __restrict__ bias,
    float* __restrict__ out)
{
    __shared__ __align__(16) float Asr[2][BM * LDR];
    __shared__ __align__(16) int   Bsr[2][BN * LDR];

    const int tid  = threadIdx.x;
    const int lane = tid & 31;
    const int wave = tid >> 5;
    const int n0 = blockIdx.x * BN;
    const int m0 = blockIdx.y * BM;
    const int wave_m  = wave >> 2;
    const int wave_n  = wave & 3;
    const int halfsel = lane >> 4;
    const int l16     = lane & 15;

    float sreg[2];
#pragma unroll
    for (int ni = 0; ni < 2; ++ni)
        sreg[ni] = scale[n0 + wave_n * 32 + ni * 16 + l16];

    v8f acc[4][2] = {};

    auto stage = [&](int kt, int buf) {
#pragma unroll
        for (int i = 0; i < 4; ++i) {
            int idx = tid + i * 256;
            int row = idx >> 3;
            int c4  = (idx & 7) * 4;
            const float* gA = &x[(size_t)(m0 + row) * IN_F + kt + c4];
            float*       lA = &Asr[buf][row * LDR + c4];
            const int*   gB = &qweight[(size_t)(n0 + row) * IN_F + kt + c4];
            int*         lB = &Bsr[buf][row * LDR + c4];
#if HAVE_ASYNC_LDS
            __builtin_amdgcn_global_load_async_to_lds_b128(AS1P(gA), AS3P(lA), 0, 0);
            __builtin_amdgcn_global_load_async_to_lds_b128(AS1P(gB), AS3P(lB), 0, 0);
#else
            *reinterpret_cast<float4*>(lA) = *reinterpret_cast<const float4*>(gA);
            *reinterpret_cast<int4*>(lB)   = *reinterpret_cast<const int4*>(gB);
#endif
        }
    };

    const int NK = IN_F / BK;
    stage(0, 0);

    for (int t = 0; t < NK; ++t) {
        const int cur = t & 1;
        __syncthreads();
        if (t + 1 < NK) { stage((t + 1) * BK, 1 - cur); async_wait_le8(); }
        else            { async_wait_0(); }
        __syncthreads();

        v16h afrag[4];
#pragma unroll
        for (int mi = 0; mi < 4; ++mi) {
            const float* base =
                &Asr[cur][(wave_m * 64 + mi * 16 + l16) * LDR + halfsel * 8];
            float pa[16];
            *reinterpret_cast<float4*>(pa + 0)  = *reinterpret_cast<const float4*>(base + 0);
            *reinterpret_cast<float4*>(pa + 4)  = *reinterpret_cast<const float4*>(base + 4);
            *reinterpret_cast<float4*>(pa + 8)  = *reinterpret_cast<const float4*>(base + 16);
            *reinterpret_cast<float4*>(pa + 12) = *reinterpret_cast<const float4*>(base + 20);
            v16h a;
#pragma unroll
            for (int e = 0; e < 16; ++e) a[e] = (_Float16)pa[e];
            afrag[mi] = a;
        }
        v16h bfrag[2];
#pragma unroll
        for (int ni = 0; ni < 2; ++ni) {
            const int* base =
                &Bsr[cur][(wave_n * 32 + ni * 16 + l16) * LDR + halfsel * 8];
            int qa[16];
            *reinterpret_cast<int4*>(qa + 0)  = *reinterpret_cast<const int4*>(base + 0);
            *reinterpret_cast<int4*>(qa + 4)  = *reinterpret_cast<const int4*>(base + 4);
            *reinterpret_cast<int4*>(qa + 8)  = *reinterpret_cast<const int4*>(base + 16);
            *reinterpret_cast<int4*>(qa + 12) = *reinterpret_cast<const int4*>(base + 20);
            const float s = sreg[ni];
            v16h b;
#pragma unroll
            for (int e = 0; e < 16; ++e)
                b[e] = (_Float16)((float)(qa[e] - 8) * s);
            bfrag[ni] = b;
        }
#pragma unroll
        for (int mi = 0; mi < 4; ++mi)
#pragma unroll
            for (int ni = 0; ni < 2; ++ni)
                acc[mi][ni] = __builtin_amdgcn_wmma_f32_16x16x32_f16(
                    false, afrag[mi], false, bfrag[ni],
                    (short)0, acc[mi][ni], false, false);
    }

#pragma unroll
    for (int ni = 0; ni < 2; ++ni) {
        int col = n0 + wave_n * 32 + ni * 16 + l16;
        float bv = bias[col];
#pragma unroll
        for (int mi = 0; mi < 4; ++mi) {
            int rbase = m0 + wave_m * 64 + mi * 16 + halfsel * 8;
#pragma unroll
            for (int r = 0; r < 8; ++r)
                out[(size_t)(rbase + r) * OUT_F + col] = acc[mi][ni][r] + bv;
        }
    }
}

// ============================================================================
extern "C" void kernel_launch(void* const* d_in, const int* in_sizes, int n_in,
                              void* d_out, int out_size, void* d_ws, size_t ws_size,
                              hipStream_t stream) {
    const float* x  = (const float*)d_in[0];
    const int*   qw = (const int*)d_in[1];
    const float* sc = (const float*)d_in[2];
    const float* bs = (const float*)d_in[3];
    float* out = (float*)d_out;

    const size_t needA = (size_t)TOKENS * IN_F * sizeof(_Float16);  // 64 MiB
    const size_t needB = (size_t)OUT_F  * IN_F * sizeof(_Float16);  // 86 MiB

    dim3 grid(OUT_F / BN, TOKENS / BM);   // 86 x 64 workgroups

    if (ws_size >= needA + needB) {
        _Float16* xh = (_Float16*)d_ws;
        _Float16* wh = (_Float16*)((char*)d_ws + needA);
        cvt_x_f16<<<dim3((TOKENS * IN_F) / (256 * 8)), dim3(256), 0, stream>>>(x, xh);
        dequant_w_f16<<<dim3(OUT_F), dim3(256), 0, stream>>>(qw, sc, wh);
        gemm_f16_wmma<<<grid, dim3(256), 0, stream>>>(xh, wh, bs, out);
    } else {
        linear4bit_wmma_fused<<<grid, dim3(256), 0, stream>>>(x, qw, sc, bs, out);
    }
}
